// QuantumFluctuationAttention_31653908971969
// MI455X (gfx1250) — matrix operand order repaired and checked by reading, not compiled, not run on hardware
//
#include <hip/hip_runtime.h>
#include <hip/hip_bf16.h>

// ---------------------------------------------------------------------------
// QuantumFluctuationAttention for MI455X (gfx1250, wave32, WMMA)
//   B=4, N=2048, DIM=1024, H=16, DH=64
// Pipeline (all bf16 WMMA with f32 accumulation):
//   0) f32 -> bf16 conversions of x, qkv_w, proj_w
//   1) QKV GEMM (8192x3072x1024) + bias + q-noise; scatter q,k -> [B,H,N,DH],
//      v -> transposed [B,H,DH,N] so AV B-fragments are contiguous loads
//   2) flash attention per (b,h): online softmax, P re-layout through LDS
//   3) projection GEMM (8192x1024x1024) + bias -> f32 output
// GEMM waves use 32x64 register tiles (2x4 WMMA accumulators);
// __launch_bounds__(256) relaxes the VGPR budget so the tile stays in
// registers (round-2 build spilled under the default 1024-thread assumption).
// ---------------------------------------------------------------------------

typedef __bf16 bf16_t;
typedef __attribute__((ext_vector_type(16))) __bf16 v16bf;
typedef __attribute__((ext_vector_type(8)))  __bf16 v8bf;
typedef __attribute__((ext_vector_type(8)))  float  v8f;

#define NB    4
#define NSEQ  2048
#define DIM   1024
#define NH    16
#define DHD   64
#define MROWS (NB * NSEQ)      // 8192
#define QKVN  (3 * DIM)        // 3072
#define BHND  (NB * NH * NSEQ * DHD)  // 8388608 elements per q/k/v tensor

// ---------------------------------------------------------------------------
// WMMA wrapper: D = A(16x32 bf16) x B(32x16 bf16) + C(16x16 f32)
// ---------------------------------------------------------------------------
__device__ __forceinline__ v8f wmma_bf16f32(v16bf a, v16bf b, v8f c) {
  return __builtin_amdgcn_wmma_f32_16x16x32_bf16(
      /*neg_a=*/false, a, /*neg_b=*/false, b,
      /*c_mod=*/(short)0, c, /*reuse_a=*/false, /*reuse_b=*/false);
}

__device__ __forceinline__ v8f zero_v8f() {
  v8f z = {0.f, 0.f, 0.f, 0.f, 0.f, 0.f, 0.f, 0.f};
  return z;
}

// ---------------------------------------------------------------------------
// A-fragment (16x32, MxK) loader from row-major [*, ld] bf16.
// ISA layout: lane l holds row m=l&15; elements 0..7 -> K = koff+0..7,
// elements 8..15 -> K = koff+16..23, koff = (l<16) ? 0 : 8.
// ---------------------------------------------------------------------------
__device__ __forceinline__ v16bf load_frag_a(const bf16_t* __restrict__ base,
                                             int row0, int k0, int ld) {
  const int lane = threadIdx.x & 31;
  const int m    = lane & 15;
  const int koff = (lane < 16) ? 0 : 8;
  const bf16_t* p = base + (size_t)(row0 + m) * ld + (k0 + koff);
  v8bf lo = *reinterpret_cast<const v8bf*>(p);       // K = koff + 0..7
  v8bf hi = *reinterpret_cast<const v8bf*>(p + 16);  // K = koff + 16..23
  v16bf a;
#pragma unroll
  for (int i = 0; i < 8; ++i) { a[i] = lo[i]; a[i + 8] = hi[i]; }
  return a;
}

// ---------------------------------------------------------------------------
// B-fragment (32x16, KxN) loader where B[k][n] = W[n0+n][k0+k] and W is
// row-major [*, ld] (i.e. computes against W^T -- exactly x @ w.T form).
// ISA layout: lane l holds column n=l&15; elements 0..7 -> K = koff+0..7,
// elements 8..15 -> K = koff+16..23, koff = (l<16) ? 0 : 8.
// ---------------------------------------------------------------------------
__device__ __forceinline__ v16bf load_frag_bT(const bf16_t* __restrict__ base,
                                              int n0, int k0, int ld) {
  const int lane = threadIdx.x & 31;
  const int n    = lane & 15;
  const int koff = (lane < 16) ? 0 : 8;
  const bf16_t* p = base + (size_t)(n0 + n) * ld + (k0 + koff);
  v8bf lo = *reinterpret_cast<const v8bf*>(p);
  v8bf hi = *reinterpret_cast<const v8bf*>(p + 16);
  v16bf b;
#pragma unroll
  for (int i = 0; i < 8; ++i) { b[i] = lo[i]; b[i + 8] = hi[i]; }
  return b;
}

// ---------------------------------------------------------------------------
// Kernel 0: f32 -> bf16 conversion
// ---------------------------------------------------------------------------
__global__ __launch_bounds__(256) void f32_to_bf16_kernel(
    const float* __restrict__ in, bf16_t* __restrict__ out, int n) {
  int i = blockIdx.x * blockDim.x + threadIdx.x;
  if (i < n) out[i] = (bf16_t)in[i];
}

// ---------------------------------------------------------------------------
// Shared GEMM body: wave computes 32x64 (2x4 WMMA accumulators).
// Block = 256 threads = 8 waves (4 row-waves x 2 col-waves) -> 128x128 tile.
// ---------------------------------------------------------------------------
__device__ __forceinline__ void gemm_wave_32x64(const bf16_t* __restrict__ X,
                                                const bf16_t* __restrict__ W,
                                                int row0, int col0, int kdim,
                                                v8f acc[2][4]) {
#pragma unroll
  for (int i = 0; i < 2; ++i)
#pragma unroll
    for (int j = 0; j < 4; ++j) acc[i][j] = zero_v8f();

  const int lane = threadIdx.x & 31;
  const bf16_t* pfa = X + (size_t)(row0 + (lane & 15)) * kdim + 128;

  for (int k0 = 0; k0 < kdim; k0 += 32) {
    // speculative prefetch of the A tile 4 K-steps ahead (global_prefetch_b8)
    __builtin_prefetch(pfa + k0, 0, 1);
    v16bf a0 = load_frag_a(X, row0,      k0, kdim);
    v16bf a1 = load_frag_a(X, row0 + 16, k0, kdim);
    v16bf b0 = load_frag_bT(W, col0,      k0, kdim);
    v16bf b1 = load_frag_bT(W, col0 + 16, k0, kdim);
    v16bf b2 = load_frag_bT(W, col0 + 32, k0, kdim);
    v16bf b3 = load_frag_bT(W, col0 + 48, k0, kdim);
    acc[0][0] = wmma_bf16f32(a0, b0, acc[0][0]);
    acc[0][1] = wmma_bf16f32(a0, b1, acc[0][1]);
    acc[0][2] = wmma_bf16f32(a0, b2, acc[0][2]);
    acc[0][3] = wmma_bf16f32(a0, b3, acc[0][3]);
    acc[1][0] = wmma_bf16f32(a1, b0, acc[1][0]);
    acc[1][1] = wmma_bf16f32(a1, b1, acc[1][1]);
    acc[1][2] = wmma_bf16f32(a1, b2, acc[1][2]);
    acc[1][3] = wmma_bf16f32(a1, b3, acc[1][3]);
  }
}

// ---------------------------------------------------------------------------
// Kernel 1: QKV GEMM. Epilogue adds bias, injects gaussian noise into q, and
// scatters q,k ([B,H,N,DH]) and v (transposed [B,H,DH,N]).
// ---------------------------------------------------------------------------
__global__ __launch_bounds__(256) void qkv_gemm_kernel(
    const bf16_t* __restrict__ X,     // [8192,1024]
    const bf16_t* __restrict__ W,     // [3072,1024]
    const float*  __restrict__ bias,  // [3072]
    const float*  __restrict__ eps,   // [B,H,N,DH]
    const float*  __restrict__ temperature,
    bf16_t* __restrict__ Qo,          // [B,H,N,DH]
    bf16_t* __restrict__ Ko,          // [B,H,N,DH]
    bf16_t* __restrict__ Vt) {        // [B,H,DH,N]
  const int wave = threadIdx.x >> 5;
  const int lane = threadIdx.x & 31;
  const int wr = wave >> 1, wc = wave & 1;
  const int row0 = blockIdx.y * 128 + wr * 32;
  const int col0 = blockIdx.x * 128 + wc * 64;

  v8f acc[2][4];
  gemm_wave_32x64(X, W, row0, col0, DIM, acc);

  const float sig = 1.0f / (1.0f + __expf(-temperature[0]));
  const int n  = lane & 15;
  const int mb = (lane < 16) ? 0 : 8;
#pragma unroll
  for (int mi = 0; mi < 2; ++mi) {
#pragma unroll
    for (int ni = 0; ni < 4; ++ni) {
#pragma unroll
      for (int r = 0; r < 8; ++r) {
        const int row = row0 + mi * 16 + mb + r;   // 0..8191
        const int col = col0 + ni * 16 + n;        // 0..3071
        float val = acc[mi][ni][r] + bias[col];
        const int t    = col >> 10;                // 0=q 1=k 2=v
        const int rem  = col & 1023;
        const int h    = rem >> 6;
        const int d    = rem & 63;
        const int b    = row >> 11;                // row / NSEQ
        const int nseq = row & 2047;
        const size_t bh = (size_t)(b * NH + h);
        if (t == 0) {
          const size_t idx = (bh * NSEQ + nseq) * DHD + d;
          val += eps[idx] * sig;
          Qo[idx] = (bf16_t)val;
        } else if (t == 1) {
          Ko[(bh * NSEQ + nseq) * DHD + d] = (bf16_t)val;
        } else {
          Vt[(bh * DHD + d) * NSEQ + nseq] = (bf16_t)val;   // transposed
        }
      }
    }
  }
}

// ---------------------------------------------------------------------------
// Kernel 2: flash attention. grid = (N/128, B*H), block = 256 (8 waves).
// Each wave owns 16 query rows: Q kept as two A-fragments (K=64), keys
// streamed 32 at a time (two 16-wide S tiles), online softmax with
// half-wave shfl_xor row reductions, P re-laid out C/D->A via per-wave LDS,
// O accumulated as 4 f32 fragments over DH=64.
// ---------------------------------------------------------------------------
__global__ __launch_bounds__(256) void flash_attn_kernel(
    const bf16_t* __restrict__ Qd,
    const bf16_t* __restrict__ Kd,
    const bf16_t* __restrict__ Vt,
    bf16_t* __restrict__ Oo) {   // [B*N, DIM]
  const int bh   = blockIdx.y;
  const int wave = threadIdx.x >> 5;
  const int lane = threadIdx.x & 31;
  const bf16_t* q  = Qd + (size_t)bh * NSEQ * DHD;
  const bf16_t* k  = Kd + (size_t)bh * NSEQ * DHD;
  const bf16_t* vt = Vt + (size_t)bh * DHD * NSEQ;
  const int q0 = blockIdx.x * 128 + wave * 16;

  __shared__ bf16_t pshm[8][16 * 32];
  bf16_t* myshm = pshm[wave];

  const v16bf qa0 = load_frag_a(q, q0, 0, DHD);
  const v16bf qa1 = load_frag_a(q, q0, 32, DHD);

  v8f o[4];
#pragma unroll
  for (int dt = 0; dt < 4; ++dt) o[dt] = zero_v8f();
  float mrow[8], lrow[8];
#pragma unroll
  for (int r = 0; r < 8; ++r) { mrow[r] = -3.0e38f; lrow[r] = 0.0f; }

  const float scale = 0.125f;  // 1/sqrt(64)
  const int cn = lane & 15;    // column within 16 / half-row select below

  for (int kb = 0; kb < NSEQ; kb += 32) {
    // S = Q x K^T for 32 keys (two 16-key tiles), f32 accum
    v8f s0 = zero_v8f(), s1 = zero_v8f();
    {
      v16bf kb00 = load_frag_bT(k, kb, 0, DHD);
      v16bf kb01 = load_frag_bT(k, kb, 32, DHD);
      s0 = wmma_bf16f32(qa0, kb00, s0);
      s0 = wmma_bf16f32(qa1, kb01, s0);
      v16bf kb10 = load_frag_bT(k, kb + 16, 0, DHD);
      v16bf kb11 = load_frag_bT(k, kb + 16, 32, DHD);
      s1 = wmma_bf16f32(qa0, kb10, s1);
      s1 = wmma_bf16f32(qa1, kb11, s1);
    }

    // Online softmax per owned row (r -> row r + 8*half); row's 16 columns
    // live in one 16-lane half, so xor masks 1..8 reduce exactly that row.
    float alpha[8];
#pragma unroll
    for (int r = 0; r < 8; ++r) {
      float v = fmaxf(s0[r], s1[r]);
#pragma unroll
      for (int mask = 1; mask <= 8; mask <<= 1)
        v = fmaxf(v, __shfl_xor(v, mask, 32));
      const float mnew = fmaxf(mrow[r], v * scale);
      alpha[r] = __expf(mrow[r] - mnew);
      const float p0 = __expf(s0[r] * scale - mnew);
      const float p1 = __expf(s1[r] * scale - mnew);
      float rs = p0 + p1;
#pragma unroll
      for (int mask = 1; mask <= 8; mask <<= 1)
        rs += __shfl_xor(rs, mask, 32);
      lrow[r] = lrow[r] * alpha[r] + rs;
      mrow[r] = mnew;
      // stage P (16x32) to this wave's LDS tile, row-major
      const int m = r + ((lane < 16) ? 0 : 8);
      myshm[m * 32 + cn]      = (bf16_t)p0;
      myshm[m * 32 + 16 + cn] = (bf16_t)p1;
    }

    // rescale O by alpha (accumulator rows align with S rows lane-wise)
#pragma unroll
    for (int dt = 0; dt < 4; ++dt)
#pragma unroll
      for (int r = 0; r < 8; ++r) o[dt][r] *= alpha[r];

    // reload P in A-fragment layout (same-wave LDS ops are in-order)
    v16bf pa;
    {
      const int m    = lane & 15;
      const int koff = (lane < 16) ? 0 : 8;
#pragma unroll
      for (int e = 0; e < 16; ++e) {
        const int kk = koff + ((e < 8) ? e : e + 8);
        pa[e] = myshm[m * 32 + kk];
      }
    }

    // O += P x V  (V^T is [DH, N] row-major -> contiguous B-fragment loads)
#pragma unroll
    for (int dt = 0; dt < 4; ++dt) {
      v16bf vb = load_frag_bT(vt, dt * 16, kb, NSEQ);
      o[dt] = wmma_bf16f32(pa, vb, o[dt]);
    }
  }

  // normalize and store to [B*N, DIM] with col = h*64 + d
  const int b = bh / NH, h = bh % NH;
  const int mb = (lane < 16) ? 0 : 8;
#pragma unroll
  for (int dt = 0; dt < 4; ++dt) {
#pragma unroll
    for (int r = 0; r < 8; ++r) {
      const int row = q0 + mb + r;
      const int col = h * DHD + dt * 16 + cn;
      Oo[((size_t)b * NSEQ + row) * DIM + col] = (bf16_t)(o[dt][r] / lrow[r]);
    }
  }
}

// ---------------------------------------------------------------------------
// Kernel 3: output projection GEMM (8192x1024x1024) + bias -> f32 out
// ---------------------------------------------------------------------------
__global__ __launch_bounds__(256) void proj_gemm_kernel(
    const bf16_t* __restrict__ A,   // [8192,1024]
    const bf16_t* __restrict__ W,   // [1024,1024]
    const float*  __restrict__ bias,
    float* __restrict__ out) {      // [8192,1024]
  const int wave = threadIdx.x >> 5;
  const int lane = threadIdx.x & 31;
  const int wr = wave >> 1, wc = wave & 1;
  const int row0 = blockIdx.y * 128 + wr * 32;
  const int col0 = blockIdx.x * 128 + wc * 64;

  v8f acc[2][4];
  gemm_wave_32x64(A, W, row0, col0, DIM, acc);

  const int n  = lane & 15;
  const int mb = (lane < 16) ? 0 : 8;
#pragma unroll
  for (int mi = 0; mi < 2; ++mi)
#pragma unroll
    for (int ni = 0; ni < 4; ++ni)
#pragma unroll
      for (int r = 0; r < 8; ++r) {
        const int row = row0 + mi * 16 + mb + r;
        const int col = col0 + ni * 16 + n;
        out[(size_t)row * DIM + col] = acc[mi][ni][r] + bias[col];
      }
}

// ---------------------------------------------------------------------------
// Host launcher
// ---------------------------------------------------------------------------
extern "C" void kernel_launch(void* const* d_in, const int* in_sizes, int n_in,
                              void* d_out, int out_size, void* d_ws, size_t ws_size,
                              hipStream_t stream) {
  const float* x      = (const float*)d_in[0];   // [B,N,DIM]
  const float* qkv_w  = (const float*)d_in[1];   // [3*DIM, DIM]
  const float* qkv_b  = (const float*)d_in[2];   // [3*DIM]
  const float* proj_w = (const float*)d_in[3];   // [DIM, DIM]
  const float* proj_b = (const float*)d_in[4];   // [DIM]
  const float* temp   = (const float*)d_in[5];   // [1]
  const float* eps    = (const float*)d_in[6];   // [B,H,N,DH]
  float* out = (float*)d_out;

  // workspace carve-up (bf16 elements); total ~88 MB
  bf16_t* x_bf    = (bf16_t*)d_ws;
  bf16_t* qkvw_bf = x_bf    + (size_t)MROWS * DIM;
  bf16_t* projw_bf= qkvw_bf + (size_t)QKVN  * DIM;
  bf16_t* q_bf    = projw_bf+ (size_t)DIM   * DIM;
  bf16_t* k_bf    = q_bf    + (size_t)BHND;
  bf16_t* vt_bf   = k_bf    + (size_t)BHND;
  bf16_t* attn_bf = vt_bf   + (size_t)BHND;

  // 0) conversions
  {
    int n = MROWS * DIM;
    f32_to_bf16_kernel<<<(n + 255) / 256, 256, 0, stream>>>(x, x_bf, n);
    n = QKVN * DIM;
    f32_to_bf16_kernel<<<(n + 255) / 256, 256, 0, stream>>>(qkv_w, qkvw_bf, n);
    n = DIM * DIM;
    f32_to_bf16_kernel<<<(n + 255) / 256, 256, 0, stream>>>(proj_w, projw_bf, n);
  }

  // 1) QKV GEMM + bias + noise + scatter
  {
    dim3 grid(QKVN / 128, MROWS / 128);  // 24 x 64
    qkv_gemm_kernel<<<grid, 256, 0, stream>>>(x_bf, qkvw_bf, qkv_b, eps, temp,
                                              q_bf, k_bf, vt_bf);
  }

  // 2) flash attention
  {
    dim3 grid(NSEQ / 128, NB * NH);      // 16 x 64
    flash_attn_kernel<<<grid, 256, 0, stream>>>(q_bf, k_bf, vt_bf, attn_bf);
  }

  // 3) output projection
  {
    dim3 grid(DIM / 128, MROWS / 128);   // 8 x 64
    proj_gemm_kernel<<<grid, 256, 0, stream>>>(attn_bf, projw_bf, proj_b, out);
  }
}
